// SparseEncoderLayer_33964601376968
// MI455X (gfx1250) — matrix operand order repaired and checked by reading, not compiled, not run on hardware
//
#include <hip/hip_runtime.h>
#include <hip/hip_bf16.h>

// ---------------------------------------------------------------------------
// Types
// ---------------------------------------------------------------------------
typedef __attribute__((ext_vector_type(16))) __bf16 v16bf;
typedef __attribute__((ext_vector_type(8)))  __bf16 v8bf;
typedef __attribute__((ext_vector_type(8)))  float  v8f;
typedef __attribute__((ext_vector_type(4)))  unsigned int u32x4;
typedef __attribute__((ext_vector_type(8)))  int          i32x8;
typedef __attribute__((ext_vector_type(4)))  int          i32x4;

// Problem constants (match reference)
#define BATCH   4
#define NSEQ    4096
#define DMODEL  1024
#define NHEAD   16
#define DHEAD   64
#define DFF     4096
#define BN      (BATCH * NSEQ)          // 16384 rows
#define LOCALW  32

// ---------------------------------------------------------------------------
// Tensor Data Mover: 2D tile DMA global -> LDS with hardware row padding.
// D# layout per CDNA5 ISA ch.8. Rows of `row_elems` bf16 are written to LDS
// back-to-back plus a 16B pad after each 64B row -> 40-element LDS stride.
// ---------------------------------------------------------------------------
__device__ __forceinline__ unsigned lds_off_u32(const void* p)
{
    // Generic pointers into LDS carry the LDS byte offset in their low 32 bits
    // (ADDR[31:0] per the CDNA5 aperture rules).
    return (unsigned)(unsigned long long)p;
}

__device__ __forceinline__ void tdm_load_tile_2d(
    const void* gptr,          // global byte address of tile start
    unsigned    ldsoff,        // LDS byte offset of tile destination
    unsigned    tile_rows,     // tile_dim1 (rows)
    unsigned    row_stride_el) // tensor_dim0_stride in elements (global row pitch)
{
    const unsigned long long ga = (unsigned long long)gptr;
    u32x4 g0;
    g0[0] = 1u;                                            // count=1, user mode
    g0[1] = ldsoff;                                        // lds_addr
    g0[2] = (unsigned)ga;                                  // global_addr[31:0]
    g0[3] = (unsigned)((ga >> 32) & 0x01FFFFFFull)         // global_addr[56:32]
          | (2u << 30);                                    // type = 2 ("image")

    i32x8 g1;
    // data_size=2B (1<<16), pad_enable (1<<20),
    // pad_interval=3 -> every 16 dwords (one 64B row), pad_amount=3 -> 4 dwords (16B)
    g1[0] = (int)((1u << 16) | (1u << 20) | (3u << 22) | (3u << 25));
    g1[1] = (int)(32u << 16);                              // tensor_dim0[15:0]=32
    g1[2] = (int)((tile_rows & 0xFFFFu) << 16);            // tensor_dim1[15:0]=rows
    g1[3] = (int)(32u << 16);                              // tile_dim0=32
    g1[4] = (int)(tile_rows & 0xFFFFu);                    // tile_dim1=rows, tile_dim2=0
    g1[5] = (int)row_stride_el;                            // tensor_dim0_stride[31:0]
    g1[6] = 0;                                             // stride hi / dim1_stride lo
    g1[7] = 0;                                             // dim1_stride hi (unused, 2D)

    const i32x4 z4 = {0, 0, 0, 0};
#if defined(__clang_major__) && (__clang_major__ >= 23)
    const i32x8 z8 = {0, 0, 0, 0, 0, 0, 0, 0};
    __builtin_amdgcn_tensor_load_to_lds(g0, g1, z4, z4, z8, 0);
#else
    __builtin_amdgcn_tensor_load_to_lds(g0, g1, z4, z4, 0);
#endif
}

// ---------------------------------------------------------------------------
// Weight convert + transpose: W[K][N] f32 -> Wt[N][K] bf16   (LDS 32x33 tile)
// ---------------------------------------------------------------------------
__global__ __launch_bounds__(256) void convert_transpose_kernel(
    const float* __restrict__ W, __bf16* __restrict__ Wt, int K, int N)
{
    __shared__ float tile[32][33];
    const int tx = threadIdx.x & 31;
    const int ty = threadIdx.x >> 5;          // 0..7
    const int n0 = blockIdx.x * 32;
    const int k0 = blockIdx.y * 32;
    #pragma unroll
    for (int i = 0; i < 4; ++i) {
        const int kk = ty + i * 8;
        tile[kk][tx] = W[(size_t)(k0 + kk) * N + n0 + tx];
    }
    __syncthreads();
    #pragma unroll
    for (int i = 0; i < 4; ++i) {
        const int nn = ty + i * 8;
        Wt[(size_t)(n0 + nn) * K + k0 + tx] = (__bf16)tile[tx][nn];
    }
}

// ---------------------------------------------------------------------------
// LayerNorm (D=1024), f32 in -> bf16 out. One 256-thread block per row.
// ---------------------------------------------------------------------------
__global__ __launch_bounds__(256) void layernorm_to_bf16(
    const float* __restrict__ x, const float* __restrict__ g,
    const float* __restrict__ be, __bf16* __restrict__ out)
{
    __shared__ float red[256];
    const int tid = threadIdx.x;
    const size_t row = blockIdx.x;
    const float4 v = ((const float4*)(x + row * DMODEL))[tid];

    red[tid] = v.x + v.y + v.z + v.w;
    __syncthreads();
    for (int off = 128; off > 0; off >>= 1) {
        if (tid < off) red[tid] += red[tid + off];
        __syncthreads();
    }
    const float mu = red[0] * (1.0f / (float)DMODEL);
    __syncthreads();

    const float dx = v.x - mu, dy = v.y - mu, dz = v.z - mu, dw = v.w - mu;
    red[tid] = dx * dx + dy * dy + dz * dz + dw * dw;
    __syncthreads();
    for (int off = 128; off > 0; off >>= 1) {
        if (tid < off) red[tid] += red[tid + off];
        __syncthreads();
    }
    const float rstd = rsqrtf(red[0] * (1.0f / (float)DMODEL) + 1e-5f);

    const int c = tid * 4;
    const float4 gv = ((const float4*)g)[tid];
    const float4 bv = ((const float4*)be)[tid];
    __bf16* op = out + row * DMODEL + c;
    op[0] = (__bf16)(dx * rstd * gv.x + bv.x);
    op[1] = (__bf16)(dy * rstd * gv.y + bv.y);
    op[2] = (__bf16)(dz * rstd * gv.z + bv.z);
    op[3] = (__bf16)(dw * rstd * gv.w + bv.w);
}

// ---------------------------------------------------------------------------
// WMMA GEMM: C[M,N] = A[M,K](bf16,row) x Bt[N,K](bf16 transposed weights)
//   WG: 256 thr = 8 waves (2x4), WG tile 128x256, wave tile 64x64
//   Tiles staged via TDM (tensor_load_to_lds) with double-buffered LDS;
//   DMA of tile k+1 overlaps the 16 v_wmma_f32_16x16x32_bf16 of tile k.
//   MODE 0: outB = bf16(acc)                         (QKV projections)
//   MODE 1: outF = resid + acc + bias[n]             (WO / W2 + residual)
//   MODE 2: outB = bf16(gelu_exact(acc + bias[n]))   (W1 + GELU)
// ---------------------------------------------------------------------------
#define LDS_STRIDE 40   // 32 data + 4-dword TDM pad: 80B stride, 20-bank step

template <int MODE>
__global__ __launch_bounds__(256) void gemm_bf16_wmma(
    const __bf16* __restrict__ A,   // [M,K]
    const __bf16* __restrict__ Bt,  // [N,K]
    int M, int N, int K,
    const float* __restrict__ bias,   // [N] (MODE 1,2)
    const float* __restrict__ resid,  // [M,N] (MODE 1)
    float* __restrict__ outF,         // (MODE 1)
    __bf16* __restrict__ outB)        // (MODE 0,2)
{
    __shared__ __align__(16) __bf16 As[2][128 * LDS_STRIDE];
    __shared__ __align__(16) __bf16 Bs[2][256 * LDS_STRIDE];

    const int tid   = threadIdx.x;
    const int wave  = tid >> 5;
    const int lane  = tid & 31;
    const int half  = lane >> 4;       // 0 or 1
    const int lr    = lane & 15;
    const int waveM = wave >> 2;       // 0..1
    const int waveN = wave & 3;        // 0..3
    const int blockM = blockIdx.y * 128;
    const int blockN = blockIdx.x * 256;

    const v8f vzero = {0.f, 0.f, 0.f, 0.f, 0.f, 0.f, 0.f, 0.f};
    v8f acc[4][4];
    #pragma unroll
    for (int i = 0; i < 4; ++i)
        #pragma unroll
        for (int j = 0; j < 4; ++j) acc[i][j] = vzero;

    const int kTiles = K >> 5;

    // Prologue: DMA tile 0 into buffer 0 (one issue per WG, from wave 0).
    if (wave == 0) {
        tdm_load_tile_2d(A  + (size_t)blockM * K, lds_off_u32(&As[0][0]), 128, (unsigned)K);
        tdm_load_tile_2d(Bt + (size_t)blockN * K, lds_off_u32(&Bs[0][0]), 256, (unsigned)K);
    }

    for (int kt = 0; kt < kTiles; ++kt) {
        const int cur = kt & 1;
        const int nxt = cur ^ 1;
        if (wave == 0) {
            if (kt + 1 < kTiles) {
                const int k1 = (kt + 1) << 5;
                tdm_load_tile_2d(A  + (size_t)blockM * K + k1, lds_off_u32(&As[nxt][0]), 128, (unsigned)K);
                tdm_load_tile_2d(Bt + (size_t)blockN * K + k1, lds_off_u32(&Bs[nxt][0]), 256, (unsigned)K);
                // in-order completion: <=2 outstanding == tile `kt` has landed
                __builtin_amdgcn_s_wait_tensorcnt((short)2);
            } else {
                __builtin_amdgcn_s_wait_tensorcnt((short)0);
            }
        }
        __syncthreads();

        // ---- fragments (ISA VGPR layouts for 16-bit A 16x32 / B 32x16) ----
        v16bf afrag[4], bfrag[4];
        #pragma unroll
        for (int i = 0; i < 4; ++i) {
            const __bf16* p = &As[cur][(waveM * 64 + i * 16 + lr) * LDS_STRIDE + half * 8];
            const v8bf lo = *(const v8bf*)p;         // K = 8*half .. +7
            const v8bf hi = *(const v8bf*)(p + 16);  // K = 16+8*half .. +7
            afrag[i] = __builtin_shufflevector(lo, hi, 0,1,2,3,4,5,6,7,8,9,10,11,12,13,14,15);
        }
        #pragma unroll
        for (int j = 0; j < 4; ++j) {
            const __bf16* p = &Bs[cur][(waveN * 64 + j * 16 + lr) * LDS_STRIDE + half * 8];
            const v8bf lo = *(const v8bf*)p;         // K = 8*half .. +7
            const v8bf hi = *(const v8bf*)(p + 16);  // K = 16+8*half .. +7
            bfrag[j] = __builtin_shufflevector(lo, hi, 0,1,2,3,4,5,6,7,8,9,10,11,12,13,14,15);
        }

        #pragma unroll
        for (int i = 0; i < 4; ++i)
            #pragma unroll
            for (int j = 0; j < 4; ++j)
                acc[i][j] = __builtin_amdgcn_wmma_f32_16x16x32_bf16(
                    false, afrag[i], false, bfrag[j], (short)0, acc[i][j], false, false);
        __syncthreads();   // tile `cur` fully consumed before it is DMA-overwritten
    }

    // ---- epilogue: C/D layout = lane(n + 16*(m>=8)), vgpr r -> m = r + 8*half ----
    #pragma unroll
    for (int i = 0; i < 4; ++i) {
        const int m0 = blockM + waveM * 64 + i * 16 + half * 8;
        #pragma unroll
        for (int j = 0; j < 4; ++j) {
            const int n = blockN + waveN * 64 + j * 16 + lr;
            const float bn = (MODE != 0) ? bias[n] : 0.0f;
            #pragma unroll
            for (int r = 0; r < 8; ++r) {
                const size_t idx = (size_t)(m0 + r) * N + n;
                const float val = acc[i][j][r];
                if (MODE == 0) {
                    outB[idx] = (__bf16)val;
                } else if (MODE == 1) {
                    outF[idx] = resid[idx] + val + bn;
                } else {
                    const float t = val + bn;
                    const float ge = 0.5f * t * (1.0f + erff(t * 0.70710678118654752f));
                    outB[idx] = (__bf16)ge;
                }
            }
        }
    }
}

// ---------------------------------------------------------------------------
// Attention: per-position diagonal dot products  dots[b,h,n] = (q.k)/8
// ---------------------------------------------------------------------------
__global__ __launch_bounds__(256) void attn_dots_kernel(
    const __bf16* __restrict__ q, const __bf16* __restrict__ k,
    float* __restrict__ dots)
{
    const int t = blockIdx.x * 256 + threadIdx.x;   // B*N*H threads
    const int h = t & (NHEAD - 1);
    const int n = (t >> 4) & (NSEQ - 1);
    const int b = t >> 16;
    const size_t base = ((size_t)b * NSEQ + n) * DMODEL + h * DHEAD;
    const __bf16* qp = q + base;
    const __bf16* kp = k + base;
    float acc = 0.0f;
    #pragma unroll
    for (int i = 0; i < DHEAD; i += 8) {
        const v8bf qv = *(const v8bf*)(qp + i);
        const v8bf kv = *(const v8bf*)(kp + i);
        #pragma unroll
        for (int c = 0; c < 8; ++c) acc += (float)qv[c] * (float)kv[c];
    }
    dots[((size_t)(b * NHEAD + h)) * NSEQ + n] = acc * 0.125f;  // DH^-0.5 = 1/8
}

// ---------------------------------------------------------------------------
// Softmax statistics per (b,h): running max + sum(exp)
// ---------------------------------------------------------------------------
__global__ __launch_bounds__(256) void attn_stats_kernel(
    const float* __restrict__ dots, float* __restrict__ stats)
{
    __shared__ float red[256];
    const int tid = threadIdx.x;
    const float* d = dots + (size_t)blockIdx.x * NSEQ;

    float m = -3.4e38f;
    for (int i = tid; i < NSEQ; i += 256) m = fmaxf(m, d[i]);
    red[tid] = m; __syncthreads();
    for (int off = 128; off > 0; off >>= 1) {
        if (tid < off) red[tid] = fmaxf(red[tid], red[tid + off]);
        __syncthreads();
    }
    m = red[0]; __syncthreads();

    float s = 0.0f;
    for (int i = tid; i < NSEQ; i += 256) s += __expf(d[i] - m);
    red[tid] = s; __syncthreads();
    for (int off = 128; off > 0; off >>= 1) {
        if (tid < off) red[tid] += red[tid + off];
        __syncthreads();
    }
    if (tid == 0) { stats[blockIdx.x * 2] = m; stats[blockIdx.x * 2 + 1] = red[0]; }
}

// ---------------------------------------------------------------------------
// Apply softmax scale to V and the strided permutation in one pass:
//   out[b, n'=loc*128+blk, :] = softmax(dots)[b,h, n=blk*32+loc] * v[b,n,:]
// ---------------------------------------------------------------------------
__global__ __launch_bounds__(256) void attn_scale_kernel(
    const __bf16* __restrict__ v, const float* __restrict__ dots,
    const float* __restrict__ stats, __bf16* __restrict__ out)
{
    const int bn = blockIdx.x;                  // b*NSEQ + n'
    const int b  = bn >> 12;
    const int np = bn & (NSEQ - 1);
    const int blk = np & 127;                   // N/LOCAL = 128
    const int loc = np >> 7;
    const int orig = blk * LOCALW + loc;

    const int c  = threadIdx.x * 4;
    const int h  = c >> 6;                      // DHEAD = 64
    const int bh = b * NHEAD + h;
    const float m = stats[bh * 2];
    const float s = stats[bh * 2 + 1];
    const float sc = __expf(dots[(size_t)bh * NSEQ + orig] - m) / s;

    const __bf16* vp = v + ((size_t)b * NSEQ + orig) * DMODEL + c;
    __bf16* op = out + (size_t)bn * DMODEL + c;
    #pragma unroll
    for (int i = 0; i < 4; ++i) op[i] = (__bf16)(sc * (float)vp[i]);
}

// ---------------------------------------------------------------------------
// Host launcher
// ---------------------------------------------------------------------------
extern "C" void kernel_launch(void* const* d_in, const int* in_sizes, int n_in,
                              void* d_out, int out_size, void* d_ws, size_t ws_size,
                              hipStream_t stream)
{
    const float* x     = (const float*)d_in[0];
    const float* ln1_g = (const float*)d_in[1];
    const float* ln1_b = (const float*)d_in[2];
    const float* wq    = (const float*)d_in[3];
    const float* wk    = (const float*)d_in[4];
    const float* wv    = (const float*)d_in[5];
    const float* wo    = (const float*)d_in[6];
    const float* bo    = (const float*)d_in[7];
    const float* ln2_g = (const float*)d_in[8];
    const float* ln2_b = (const float*)d_in[9];
    const float* w1    = (const float*)d_in[10];
    const float* b1    = (const float*)d_in[11];
    const float* w2    = (const float*)d_in[12];
    const float* b2    = (const float*)d_in[13];
    float* y = (float*)d_out;

    // Workspace carve-up (256B aligned)
    char* p = (char*)d_ws;
    auto carve = [&](size_t bytes) -> void* {
        void* r = (void*)p;
        p += (bytes + 255) & ~(size_t)255;
        return r;
    };
    __bf16* wqT  = (__bf16*)carve((size_t)DMODEL * DMODEL * 2);
    __bf16* wkT  = (__bf16*)carve((size_t)DMODEL * DMODEL * 2);
    __bf16* wvT  = (__bf16*)carve((size_t)DMODEL * DMODEL * 2);
    __bf16* woT  = (__bf16*)carve((size_t)DMODEL * DMODEL * 2);
    __bf16* w1T  = (__bf16*)carve((size_t)DFF * DMODEL * 2);     // [DFF][DMODEL]
    __bf16* w2T  = (__bf16*)carve((size_t)DMODEL * DFF * 2);     // [DMODEL][DFF]
    __bf16* hbf  = (__bf16*)carve((size_t)BN * DMODEL * 2);      // reused for h2
    __bf16* qbf  = (__bf16*)carve((size_t)BN * DMODEL * 2);
    __bf16* kbf  = (__bf16*)carve((size_t)BN * DMODEL * 2);
    __bf16* vbf  = (__bf16*)carve((size_t)BN * DMODEL * 2);
    __bf16* aobf = (__bf16*)carve((size_t)BN * DMODEL * 2);      // permuted attn out
    __bf16* ffbf = (__bf16*)carve((size_t)BN * DFF * 2);
    float*  x1   = (float*) carve((size_t)BN * DMODEL * 4);
    float*  dots = (float*) carve((size_t)BATCH * NHEAD * NSEQ * 4);
    float*  stat = (float*) carve((size_t)BATCH * NHEAD * 2 * 4);

    // 1) Weights -> bf16, transposed [N][K]
    convert_transpose_kernel<<<dim3(DMODEL/32, DMODEL/32), 256, 0, stream>>>(wq, wqT, DMODEL, DMODEL);
    convert_transpose_kernel<<<dim3(DMODEL/32, DMODEL/32), 256, 0, stream>>>(wk, wkT, DMODEL, DMODEL);
    convert_transpose_kernel<<<dim3(DMODEL/32, DMODEL/32), 256, 0, stream>>>(wv, wvT, DMODEL, DMODEL);
    convert_transpose_kernel<<<dim3(DMODEL/32, DMODEL/32), 256, 0, stream>>>(wo, woT, DMODEL, DMODEL);
    convert_transpose_kernel<<<dim3(DFF/32,    DMODEL/32), 256, 0, stream>>>(w1, w1T, DMODEL, DFF);
    convert_transpose_kernel<<<dim3(DMODEL/32, DFF/32),    256, 0, stream>>>(w2, w2T, DFF, DMODEL);

    // 2) LN1 -> bf16
    layernorm_to_bf16<<<BN, 256, 0, stream>>>(x, ln1_g, ln1_b, hbf);

    // 3) QKV projections (bf16 WMMA, TDM-staged tiles)
    const dim3 gK(DMODEL/256, BN/128);
    gemm_bf16_wmma<0><<<gK, 256, 0, stream>>>(hbf, wqT, BN, DMODEL, DMODEL, nullptr, nullptr, nullptr, qbf);
    gemm_bf16_wmma<0><<<gK, 256, 0, stream>>>(hbf, wkT, BN, DMODEL, DMODEL, nullptr, nullptr, nullptr, kbf);
    gemm_bf16_wmma<0><<<gK, 256, 0, stream>>>(hbf, wvT, BN, DMODEL, DMODEL, nullptr, nullptr, nullptr, vbf);

    // 4) Diagonal attention + softmax + permuted scale of V
    attn_dots_kernel <<<(BATCH * NSEQ * NHEAD) / 256, 256, 0, stream>>>(qbf, kbf, dots);
    attn_stats_kernel<<<BATCH * NHEAD, 256, 0, stream>>>(dots, stat);
    attn_scale_kernel<<<BATCH * NSEQ, 256, 0, stream>>>(vbf, dots, stat, aobf);

    // 5) Output projection + residual: x1 = x + ao @ wo + bo
    gemm_bf16_wmma<1><<<gK, 256, 0, stream>>>(aobf, woT, BN, DMODEL, DMODEL, bo, x, x1, nullptr);

    // 6) LN2 -> bf16 (reuse hbf)
    layernorm_to_bf16<<<BN, 256, 0, stream>>>(x1, ln2_g, ln2_b, hbf);

    // 7) FFN: gelu(h2 @ w1 + b1) -> bf16
    gemm_bf16_wmma<2><<<dim3(DFF/256, BN/128), 256, 0, stream>>>(hbf, w1T, BN, DFF, DMODEL, b1, nullptr, nullptr, ffbf);

    // 8) y = x1 + ff @ w2 + b2
    gemm_bf16_wmma<1><<<gK, 256, 0, stream>>>(ffbf, w2T, BN, DMODEL, DFF, b2, x1, y, nullptr);
}
